// MultiHeadAttention_61435212202097
// MI455X (gfx1250) — matrix operand order repaired and checked by reading, not compiled, not run on hardware
//
#include <hip/hip_runtime.h>
#include <hip/hip_bf16.h>
#include <math.h>
#include <stdint.h>

// Problem constants (fixed by the reference)
#define BATCH 2
#define SEQ   2048
#define EMB   1024
#define NHEAD 16
#define HDIM  64
#define MROWS (BATCH * SEQ)   // 4096

typedef __attribute__((ext_vector_type(16))) __bf16 v16bf;
typedef __attribute__((ext_vector_type(8)))  float  v8f;

static __device__ __forceinline__ unsigned short f32_to_bf16(float f) {
  unsigned int u = __float_as_uint(f);
  unsigned int r = u + 0x7FFFu + ((u >> 16) & 1u);   // round-to-nearest-even
  return (unsigned short)(r >> 16);
}
static __device__ __forceinline__ float bf16lo(unsigned v) {
  return __uint_as_float(v << 16);
}
static __device__ __forceinline__ float bf16hi(unsigned v) {
  return __uint_as_float(v & 0xFFFF0000u);
}
static __device__ __forceinline__ unsigned pack_bf16(float a, float b) {
  return (unsigned)f32_to_bf16(a) | ((unsigned)f32_to_bf16(b) << 16);
}

union FragU {
  uint4 u[2];
  v16bf v;
};

static __device__ __forceinline__ v16bf comb(uint4 lo, uint4 hi) {
  FragU f;
  f.u[0] = lo;
  f.u[1] = hi;
  return f.v;
}

#define LD16(p) (*(const uint4*)(p))

static __device__ __forceinline__ v8f wmma_bf16(v16bf a, v16bf b, v8f c) {
  // D = A(16x32) * B(32x16) + C(16x16), f32 accumulate
  return __builtin_amdgcn_wmma_f32_16x16x32_bf16(false, a, false, b,
                                                 (short)0, c, false, false);
}

// ---- CDNA5 async global->LDS copy (ASYNCcnt-tracked, no VGPR round trip) ----
static __device__ __forceinline__ void async_load_b128(const void* gaddr, void* lds) {
  unsigned      laddr = (unsigned)(uintptr_t)lds;        // low 32b = LDS byte addr
  unsigned long long ga = (unsigned long long)(uintptr_t)gaddr;
  asm volatile("global_load_async_to_lds_b128 %0, %1, off"
               :: "v"(laddr), "v"(ga)
               : "memory");
}
static __device__ __forceinline__ void wait_asynccnt0() {
#if __has_builtin(__builtin_amdgcn_s_wait_asynccnt)
  __builtin_amdgcn_s_wait_asynccnt(0);
#else
  asm volatile("s_wait_asynccnt 0x0" ::: "memory");
#endif
}

// ---------------------------------------------------------------------------
// fp32 -> bf16 conversion, vectorized (float4 in, 4 packed bf16 out)
// ---------------------------------------------------------------------------
__global__ void cvt_f32_bf16_v4(const float4* __restrict__ src,
                                uint2* __restrict__ dst, int n4) {
  int i = blockIdx.x * blockDim.x + threadIdx.x;
  int stride = gridDim.x * blockDim.x;
  for (; i < n4; i += stride) {
    float4 f = src[i];
    uint2 o;
    o.x = pack_bf16(f.x, f.y);
    o.y = pack_bf16(f.z, f.w);
    dst[i] = o;
  }
}

// ---------------------------------------------------------------------------
// C[m,n] = sum_k A[m,k] * W[n,k] + bias[n]   (i.e. A @ W^T + b)
// One wave computes a 32x64 tile (2 M-subtiles x 4 N-tiles, 8 accumulators).
// Two-deep register ping-pong pipeline (verified in asm: clause'd loads +
// partial s_wait_loadcnt interleaved with WMMA issue).
// mode 0: out bf16 [B,H,S,HD] | mode 1: out bf16 [B,H,HD,S] | mode 2: fp32 [M,E]
// ---------------------------------------------------------------------------
struct KChunk {
  uint4 a0l, a0h, a1l, a1h;  // A fragments for the two 16-row subtiles
  uint4 bl[4], bh[4];        // B fragments for the four 16-col tiles
};

static __device__ __forceinline__ void load_chunk(KChunk& c,
                                                  const unsigned short* arow0,
                                                  const unsigned short* arow1,
                                                  const unsigned short* wbase,
                                                  int k0) {
  c.a0l = LD16(arow0 + k0);
  c.a0h = LD16(arow0 + k0 + 16);
  c.a1l = LD16(arow1 + k0);
  c.a1h = LD16(arow1 + k0 + 16);
#pragma unroll
  for (int j = 0; j < 4; ++j) {
    const unsigned short* wp = wbase + (size_t)j * 16 * EMB + k0;
    c.bl[j] = LD16(wp);
    c.bh[j] = LD16(wp + 16);
  }
}

static __device__ __forceinline__ void mma_chunk(const KChunk& c,
                                                 v8f acc0[4], v8f acc1[4]) {
  v16bf av0 = comb(c.a0l, c.a0h);
  v16bf av1 = comb(c.a1l, c.a1h);
#pragma unroll
  for (int j = 0; j < 4; ++j) {
    v16bf bv = comb(c.bl[j], c.bh[j]);
    acc0[j] = wmma_bf16(av0, bv, acc0[j]);
    acc1[j] = wmma_bf16(av1, bv, acc1[j]);
  }
}

__global__ __launch_bounds__(256)
void gemm_xwt(const unsigned short* __restrict__ A,
              const unsigned short* __restrict__ W,
              const float* __restrict__ bias,
              unsigned short* __restrict__ out_bf,
              float* __restrict__ out_f,
              int mode) {
  const int lane = threadIdx.x & 31;
  const int l15  = lane & 15;
  const int lh   = lane >> 4;
  const int wid  = (blockIdx.x * blockDim.x + threadIdx.x) >> 5;
  const int MT2  = MROWS / 32;  // 128
  const int NT64 = EMB / 64;    // 16
  if (wid >= MT2 * NT64) return;
  const int mtile2 = wid % MT2;
  const int n64    = wid / MT2;

  const v8f zero = {0.f, 0.f, 0.f, 0.f, 0.f, 0.f, 0.f, 0.f};
  v8f acc0[4] = {zero, zero, zero, zero};
  v8f acc1[4] = {zero, zero, zero, zero};

  // Lane-relative base pointers (A and B chunks at k0 + lh*8 / +16)
  const unsigned short* arow0 = A + (size_t)(mtile2 * 32 + l15) * EMB + lh * 8;
  const unsigned short* arow1 = arow0 + (size_t)16 * EMB;
  const unsigned short* wbase = W + (size_t)(n64 * 64 + l15) * EMB + lh * 8;

  KChunk c0, c1;
  load_chunk(c0, arow0, arow1, wbase, 0);
#pragma unroll 1
  for (int k0 = 0; k0 < EMB; k0 += 64) {
    load_chunk(c1, arow0, arow1, wbase, k0 + 32);   // in flight over c0's WMMAs
    mma_chunk(c0, acc0, acc1);
    if (k0 + 64 < EMB) load_chunk(c0, arow0, arow1, wbase, k0 + 64);
    mma_chunk(c1, acc0, acc1);
  }

#pragma unroll
  for (int t = 0; t < 2; ++t) {
    v8f* acc = t ? acc1 : acc0;
#pragma unroll
    for (int j = 0; j < 4; ++j) {
      const int col = n64 * 64 + j * 16 + l15;
      const float bv = bias[col];
#pragma unroll
      for (int r = 0; r < 8; ++r) {
        const int mrow = mtile2 * 32 + t * 16 + r + lh * 8;
        const float val = acc[j][r] + bv;
        if (mode == 2) {
          out_f[(size_t)mrow * EMB + col] = val;
        } else {
          const int b = mrow >> 11;           // / SEQ
          const int s = mrow & (SEQ - 1);
          const int h = col >> 6;             // / HDIM
          const int d = col & (HDIM - 1);
          size_t idx;
          if (mode == 0) idx = (((size_t)(b * NHEAD + h) * SEQ) + s) * HDIM + d;
          else           idx = (((size_t)(b * NHEAD + h) * HDIM) + d) * SEQ + s;
          out_bf[idx] = f32_to_bf16(val);
        }
      }
    }
  }
}

// ---------------------------------------------------------------------------
// Attention: 4 waves/WG, one 16-row q-tile per wave; all 4 waves of a block
// share the same (b,h), so K / V^T tiles are staged ONCE per block into
// double-buffered LDS via GLOBAL_LOAD_ASYNC_TO_LDS_B128 (ASYNCcnt), with
// s_wait_asynccnt 0 + one barrier per tile.
//   pass 1: scores = (Q K^T)/8, masked -> private bf16 LDS slab (2 WMMA/tile)
//   pass 2: softmax: 8x ds_read_b128 per row into registers, max/exp/sum,
//           8x ds_store_b128 back (single pass over the row)
//   pass 3: O = P V from staged V^T tiles (4 WMMA/chunk)
// LDS: 4*64KB slabs + 8KB stage + 256B inv = ~270 KB of 320 KB.
// ---------------------------------------------------------------------------
#define SLAB_BYTES   (4 * 16 * SEQ * 2)          // 262144
#define STAGE_BYTES  8192
#define ATTN_SMEM    (SLAB_BYTES + STAGE_BYTES + 4 * 16 * 4)

__global__ __launch_bounds__(128)
void attn_kernel(const unsigned short* __restrict__ Q,   // [B,H,S,HD]
                 const unsigned short* __restrict__ K,   // [B,H,S,HD]
                 const unsigned short* __restrict__ VT,  // [B,H,HD,S]
                 const unsigned char* __restrict__ mask, // [B,S] (true = masked)
                 unsigned short* __restrict__ O) {       // [B*S, EMB] bf16
  extern __shared__ char smem[];
  const int tid    = threadIdx.x;
  const int lane   = tid & 31;
  const int l15    = lane & 15;
  const int lh     = lane >> 4;
  const int waveid = tid >> 5;

  unsigned short* sbuf = (unsigned short*)smem + (size_t)waveid * 16 * SEQ;
  unsigned short* kst  = (unsigned short*)(smem + SLAB_BYTES);       // [2][16*64]
  unsigned short* vst  = (unsigned short*)(smem + SLAB_BYTES);       // [2][64*32]
  float* rowinv = (float*)(smem + SLAB_BYTES + STAGE_BYTES) + waveid * 16;

  const int gw = blockIdx.x * 4 + waveid;   // 0..4095
  const int qt = gw & (SEQ / 16 - 1);       // q-tile 0..127
  const int bh = gw >> 7;                   // 0..31 (same for all 4 waves)
  const int b  = bh >> 4;
  const int h  = bh & (NHEAD - 1);

  const unsigned short* Qh = Q  + (size_t)bh * SEQ * HDIM;
  const unsigned short* Kh = K  + (size_t)bh * SEQ * HDIM;
  const unsigned short* Vh = VT + (size_t)bh * HDIM * SEQ;
  const unsigned char*  mrow = mask + (size_t)b * SEQ;

  const v8f zero = {0.f, 0.f, 0.f, 0.f, 0.f, 0.f, 0.f, 0.f};

  // Q A-fragments for this 16-row tile (K-dim = HD = 64 -> two fragments)
  const unsigned short* qp = Qh + (size_t)(qt * 16 + l15) * HDIM;
  const v16bf aq0 = comb(LD16(qp + lh * 8),      LD16(qp + lh * 8 + 16));
  const v16bf aq1 = comb(LD16(qp + 32 + lh * 8), LD16(qp + 32 + lh * 8 + 16));

  // ---- pass 1: scores -> private LDS slab; K tiles staged via async DMA ----
  auto stage_k = [&](int nt, int buf) {
    // 128 threads x one async b128: the 16x64 bf16 K tile (2KB)
    const int row = tid >> 3;
    const int dc  = (tid & 7) * 8;
    async_load_b128(Kh + (size_t)(nt * 16 + row) * HDIM + dc,
                    kst + buf * 1024 + row * 64 + dc);
  };
  auto do_s = [&](int nt, int buf) {
    const unsigned short* kb = kst + buf * 1024 + l15 * 64 + lh * 8;
    v8f c = zero;
    c = wmma_bf16(aq0, comb(LD16(kb),      LD16(kb + 16)),  c);
    c = wmma_bf16(aq1, comb(LD16(kb + 32), LD16(kb + 48)), c);
    const int col = nt * 16 + l15;
    const bool mk = mrow[col] != 0;
#pragma unroll
    for (int r = 0; r < 8; ++r) {
      float sv = c[r] * 0.125f;                 // 1/sqrt(64)
      if (mk) sv = -1.0e9f;
      sbuf[(size_t)(r + lh * 8) * SEQ + col] = f32_to_bf16(sv);
    }
  };
  stage_k(0, 0);
  wait_asynccnt0();
  __syncthreads();
#pragma unroll 1
  for (int nt = 0; nt < SEQ / 16; ++nt) {
    if (nt + 1 < SEQ / 16) stage_k(nt + 1, (nt + 1) & 1);  // DMA over WMMAs
    do_s(nt, nt & 1);
    wait_asynccnt0();
    __syncthreads();   // next-tile data visible; current buffer free to reuse
  }

  // ---- pass 2: softmax, vectorized b128 LDS, single pass per row ----
#pragma unroll 1
  for (int r = 0; r < 16; ++r) {
    uint4* row4 = (uint4*)(sbuf + (size_t)r * SEQ);   // 256 uint4 per row
    uint4 vals[8];
    float m = -3.0e38f;
#pragma unroll
    for (int u = 0; u < 8; ++u) {
      uint4 v = row4[lane + 32 * u];
      vals[u] = v;
      m = fmaxf(m, fmaxf(fmaxf(bf16lo(v.x), bf16hi(v.x)),
                         fmaxf(bf16lo(v.y), bf16hi(v.y))));
      m = fmaxf(m, fmaxf(fmaxf(bf16lo(v.z), bf16hi(v.z)),
                         fmaxf(bf16lo(v.w), bf16hi(v.w))));
    }
#pragma unroll
    for (int off = 16; off > 0; off >>= 1) m = fmaxf(m, __shfl_xor(m, off, 32));
    float sum = 0.f;
#pragma unroll
    for (int u = 0; u < 8; ++u) {
      uint4 v = vals[u];
      float e0 = __expf(bf16lo(v.x) - m), e1 = __expf(bf16hi(v.x) - m);
      float e2 = __expf(bf16lo(v.y) - m), e3 = __expf(bf16hi(v.y) - m);
      float e4 = __expf(bf16lo(v.z) - m), e5 = __expf(bf16hi(v.z) - m);
      float e6 = __expf(bf16lo(v.w) - m), e7 = __expf(bf16hi(v.w) - m);
      sum += (e0 + e1) + (e2 + e3) + (e4 + e5) + (e6 + e7);
      uint4 o;
      o.x = pack_bf16(e0, e1);
      o.y = pack_bf16(e2, e3);
      o.z = pack_bf16(e4, e5);
      o.w = pack_bf16(e6, e7);
      row4[lane + 32 * u] = o;
    }
#pragma unroll
    for (int off = 16; off > 0; off >>= 1) sum += __shfl_xor(sum, off, 32);
    if (lane == 0) rowinv[r] = 1.0f / sum;
  }

  // ---- pass 3: O = P x V; V^T tiles staged via async DMA ----
  v8f oacc[4] = {zero, zero, zero, zero};
  auto stage_v = [&](int t, int buf) {
    // 128 threads x two async b128: the 64x32 bf16 V^T tile (4KB)
    const int d  = tid >> 1;
    const int kc = (tid & 1) * 16;
    const unsigned short* gp = Vh + (size_t)d * SEQ + t * 32 + kc;
    unsigned short* lp = vst + buf * 2048 + d * 32 + kc;
    async_load_b128(gp,     lp);
    async_load_b128(gp + 8, lp + 8);
  };
  auto do_p = [&](int t, int buf) {
    const unsigned short* pp = sbuf + (size_t)l15 * SEQ + t * 32 + lh * 8;
    v16bf pa = comb(LD16(pp), LD16(pp + 16));
#pragma unroll
    for (int j = 0; j < 4; ++j) {
      const unsigned short* vb = vst + buf * 2048 + (j * 16 + l15) * 32 + lh * 8;
      oacc[j] = wmma_bf16(pa, comb(LD16(vb), LD16(vb + 16)), oacc[j]);
    }
  };
  stage_v(0, 0);
  wait_asynccnt0();
  __syncthreads();
#pragma unroll 1
  for (int t = 0; t < SEQ / 32; ++t) {
    if (t + 1 < SEQ / 32) stage_v(t + 1, (t + 1) & 1);
    do_p(t, t & 1);
    wait_asynccnt0();
    __syncthreads();
  }

#pragma unroll
  for (int j = 0; j < 4; ++j) {
#pragma unroll
    for (int r = 0; r < 8; ++r) {
      const int rr = r + lh * 8;
      const int qa = qt * 16 + rr;
      const int d  = j * 16 + l15;
      const float val = oacc[j][r] * rowinv[rr];
      O[((size_t)(b * SEQ + qa)) * EMB + h * HDIM + d] = f32_to_bf16(val);
    }
  }
}

// ---------------------------------------------------------------------------
extern "C" void kernel_launch(void* const* d_in, const int* in_sizes, int n_in,
                              void* d_out, int out_size, void* d_ws, size_t ws_size,
                              hipStream_t stream) {
  (void)in_sizes; (void)n_in; (void)out_size; (void)ws_size;
  const float*         x    = (const float*)d_in[0];
  const unsigned char* mask = (const unsigned char*)d_in[1];
  const float* Wq = (const float*)d_in[2];
  const float* bq = (const float*)d_in[3];
  const float* Wk = (const float*)d_in[4];
  const float* bk = (const float*)d_in[5];
  const float* Wv = (const float*)d_in[6];
  const float* bv = (const float*)d_in[7];
  const float* Wo = (const float*)d_in[8];
  const float* bo = (const float*)d_in[9];
  float* out = (float*)d_out;

  char* ws = (char*)d_ws;
  const size_t MB = 1u << 20;
  unsigned short* x_bf  = (unsigned short*)(ws + 0 * MB);   // 8 MB
  unsigned short* wq_bf = (unsigned short*)(ws + 8 * MB);   // 2 MB
  unsigned short* wk_bf = (unsigned short*)(ws + 10 * MB);  // 2 MB
  unsigned short* wv_bf = (unsigned short*)(ws + 12 * MB);  // 2 MB
  unsigned short* wo_bf = (unsigned short*)(ws + 14 * MB);  // 2 MB
  unsigned short* q_bf  = (unsigned short*)(ws + 16 * MB);  // 8 MB [B,H,S,D]
  unsigned short* k_bf  = (unsigned short*)(ws + 24 * MB);  // 8 MB [B,H,S,D]
  unsigned short* vt_bf = (unsigned short*)(ws + 32 * MB);  // 8 MB [B,H,D,S]
  unsigned short* o_bf  = (unsigned short*)(ws + 40 * MB);  // 8 MB [B*S,E]

  // 1) bf16 conversions (vectorized)
  cvt_f32_bf16_v4<<<2048, 256, 0, stream>>>((const float4*)x,  (uint2*)x_bf,  MROWS * EMB / 4);
  cvt_f32_bf16_v4<<<512,  256, 0, stream>>>((const float4*)Wq, (uint2*)wq_bf, EMB * EMB / 4);
  cvt_f32_bf16_v4<<<512,  256, 0, stream>>>((const float4*)Wk, (uint2*)wk_bf, EMB * EMB / 4);
  cvt_f32_bf16_v4<<<512,  256, 0, stream>>>((const float4*)Wv, (uint2*)wv_bf, EMB * EMB / 4);
  cvt_f32_bf16_v4<<<512,  256, 0, stream>>>((const float4*)Wo, (uint2*)wo_bf, EMB * EMB / 4);

  // 2) QKV projections: 2048 waves (32x64 tile each) -> 256 blocks of 256
  gemm_xwt<<<256, 256, 0, stream>>>(x_bf, wq_bf, bq, q_bf,  nullptr, 0);
  gemm_xwt<<<256, 256, 0, stream>>>(x_bf, wk_bf, bk, k_bf,  nullptr, 0);
  gemm_xwt<<<256, 256, 0, stream>>>(x_bf, wv_bf, bv, vt_bf, nullptr, 1);

  // 3) attention: 4096 q-tiles, 4 waves/block sharing async-staged K/V tiles
  attn_kernel<<<1024, 128, ATTN_SMEM, stream>>>(q_bf, k_bf, vt_bf, mask, o_bf);

  // 4) output projection -> fp32 d_out
  gemm_xwt<<<256, 256, 0, stream>>>(o_bf, wo_bf, bo, nullptr, out, 2);
}
